// CrystalDecoder_79637283602844
// MI455X (gfx1250) — matrix operand order repaired and checked by reading, not compiled, not run on hardware
//
#include <hip/hip_runtime.h>

#define NB     1024
#define NNODE  204800
#define NEDGE  1000000

typedef __attribute__((ext_vector_type(16))) __bf16        v16bf;
typedef __attribute__((ext_vector_type(8)))  float         v8f;
typedef __attribute__((ext_vector_type(4)))  unsigned int  v4u_;
typedef __attribute__((ext_vector_type(8)))  int           v8i_;
typedef __attribute__((ext_vector_type(4)))  int           v4i_;

union FragB {
  v16bf v;
  v4u_  q[2];
  unsigned short s[16];
};

__device__ __forceinline__ unsigned short f2bf(float f) {
  unsigned int u = __float_as_uint(f);
  u += 0x7FFFu + ((u >> 16) & 1u);       // round-to-nearest-even truncation
  return (unsigned short)(u >> 16);
}

__device__ __forceinline__ float gelu_f(float x) {
  return 0.5f * x * (1.0f + erff(x * 0.7071067811865475f));
}

// TDM 2D tile load: global (row-major, dim0 cols x dim1 rows, bf16) -> LDS,
// with pad_enable so each row lands at a bank-conflict-padded pitch.
// pad_code: 5 => pad every 64 DW (128 bf16 row), 4 => every 32 DW (64 bf16 row)
// amt_code: 3 => insert 4 DW (8 bf16) of padding  => pitch 136 / 72 halfwords.
__device__ __forceinline__ void tdm_load2d(unsigned lds_off, const void* gp,
                                           unsigned dim0, unsigned dim1,
                                           unsigned pad_code, unsigned amt_code) {
  unsigned long long ga = (unsigned long long)(size_t)gp;
  v4u_ g0;
  g0[0] = 1u;                                       // count=1, user descriptor
  g0[1] = lds_off;                                  // lds_addr
  g0[2] = (unsigned)(ga & 0xFFFFFFFFull);           // global_addr[31:0]
  g0[3] = ((unsigned)((ga >> 32) & 0x1FFFFFFull)) | (2u << 30);  // addr[56:32], type=2
  v8i_ g1;
  g1[0] = (int)((1u << 16) | (1u << 20) |           // data_size=2B, pad_enable
                ((pad_code & 7u) << 22) | ((amt_code & 127u) << 25));
  g1[1] = (int)((dim0 & 0xFFFFu) << 16);            // tensor_dim0 lo16 (bits 63:48)
  g1[2] = (int)(((dim0 >> 16) & 0xFFFFu) | ((dim1 & 0xFFFFu) << 16)); // dim0 hi / dim1 lo
  g1[3] = (int)(((dim1 >> 16) & 0xFFFFu) | ((dim0 & 0xFFFFu) << 16)); // dim1 hi / tile_dim0
  g1[4] = (int)(dim1 & 0xFFFFu);                    // tile_dim1, tile_dim2=0
  g1[5] = (int)dim0;                                // tensor_dim0_stride lo32
  g1[6] = 0;                                        // stride0 hi / stride1 lo
  g1[7] = 0;
  v4i_ z4 = {0, 0, 0, 0};
  v8i_ z8 = {0, 0, 0, 0, 0, 0, 0, 0};
  __builtin_amdgcn_tensor_load_to_lds(g0, g1, z4, z4, z8, 0);
}

template <int NT>
__device__ __forceinline__ float block_sum(float v, float* sh) {
  int t = threadIdx.x;
  sh[t] = v;
  __syncthreads();
  for (int s = NT >> 1; s > 0; s >>= 1) {
    if (t < s) sh[t] += sh[t + s];
    __syncthreads();
  }
  float r = sh[0];
  __syncthreads();
  return r;
}

// ---------------------------------------------------------------------------
// fp32 -> bf16 weight conversion (once per launch; deterministic)
// ---------------------------------------------------------------------------
__global__ __launch_bounds__(256) void convert_kernel(
    const float* __restrict__ ew1, const float* __restrict__ ew2,
    const float* __restrict__ nw,  const float* __restrict__ nd1,
    unsigned short* __restrict__ o1, unsigned short* __restrict__ o2,
    unsigned short* __restrict__ o3, unsigned short* __restrict__ o4) {
  int i = blockIdx.x * blockDim.x + threadIdx.x;
  if (i < 32768) o1[i] = f2bf(ew1[i]);   // ed_w1 256x128
  if (i <  8192) o2[i] = f2bf(ew2[i]);   // ed_w2 128x64
  if (i < 16384) o3[i] = f2bf(nw[i]);    // ne_w  128x128
  if (i <  8192) o4[i] = f2bf(nd1[i]);   // nd_w1 128x64
}

// ---------------------------------------------------------------------------
// z_proj = LN(GELU(z @ lp_w + lp_b)); writes f32 + bf16 copies (tiny: B=1024)
// ---------------------------------------------------------------------------
__global__ __launch_bounds__(128) void zproj_kernel(
    const float* __restrict__ z, const float* __restrict__ w,
    const float* __restrict__ b, const float* __restrict__ g,
    const float* __restrict__ be,
    float* __restrict__ zpf, unsigned short* __restrict__ zpb) {
  __shared__ float sh[128];
  __shared__ float zr[32];
  int r = blockIdx.x, j = threadIdx.x;
  if (j < 32) zr[j] = z[r * 32 + j];
  __syncthreads();
  float a = b[j];
  #pragma unroll
  for (int k = 0; k < 32; ++k) a += zr[k] * w[k * 128 + j];
  a = gelu_f(a);
  float mu = block_sum<128>(a, sh) * (1.0f / 128.0f);
  float d = a - mu;
  float var = block_sum<128>(d * d, sh) * (1.0f / 128.0f);
  float y = d * rsqrtf(var + 1e-5f) * g[j] + be[j];
  zpf[r * 128 + j] = y;
  zpb[r * 128 + j] = f2bf(y);
}

// ---------------------------------------------------------------------------
// energy + stress heads (tiny: B=1024)
// ---------------------------------------------------------------------------
__global__ __launch_bounds__(64) void heads_kernel(
    const float* __restrict__ z,
    const float* __restrict__ ew1, const float* __restrict__ eb1,
    const float* __restrict__ eg,  const float* __restrict__ ebe,
    const float* __restrict__ ew2, const float* __restrict__ eb2,
    const float* __restrict__ sw1, const float* __restrict__ sb1,
    const float* __restrict__ sg,  const float* __restrict__ sbe,
    const float* __restrict__ sw2, const float* __restrict__ sb2,
    float* __restrict__ pe, float* __restrict__ ps) {
  __shared__ float sh[64];
  __shared__ float hn[64];
  __shared__ float zr[32];
  int r = blockIdx.x, j = threadIdx.x;
  if (j < 32) zr[j] = z[r * 32 + j];
  __syncthreads();
  // energy
  float a = eb1[j];
  #pragma unroll
  for (int k = 0; k < 32; ++k) a += zr[k] * ew1[k * 64 + j];
  a = gelu_f(a);
  float mu = block_sum<64>(a, sh) * (1.0f / 64.0f);
  float d = a - mu;
  float var = block_sum<64>(d * d, sh) * (1.0f / 64.0f);
  hn[j] = d * rsqrtf(var + 1e-5f) * eg[j] + ebe[j];
  __syncthreads();
  if (j < 2) {
    float o = eb2[j];
    #pragma unroll
    for (int k = 0; k < 64; ++k) o += hn[k] * ew2[k * 2 + j];
    pe[r * 2 + j] = o;
  }
  __syncthreads();
  // stress
  a = sb1[j];
  #pragma unroll
  for (int k = 0; k < 32; ++k) a += zr[k] * sw1[k * 64 + j];
  a = gelu_f(a);
  mu = block_sum<64>(a, sh) * (1.0f / 64.0f);
  d = a - mu;
  var = block_sum<64>(d * d, sh) * (1.0f / 64.0f);
  hn[j] = d * rsqrtf(var + 1e-5f) * sg[j] + sbe[j];
  __syncthreads();
  if (j < 9) {
    float o = sb2[j];
    #pragma unroll
    for (int k = 0; k < 64; ++k) o += hn[k] * sw2[k * 9 + j];
    ps[r * 9 + j] = o;
  }
}

// ---------------------------------------------------------------------------
// EDGE kernel: per wave, a tile of 16 edges.
//   L1: [16,256]bf16 @ [256,128]bf16 (LDS via TDM) -> GELU+LN -> bf16 LDS stage
//   L2: [16,128] @ [128,64] -> GELU+LN -> f32 LDS stage -> [64,3] proj
// LDS layout: W1 @0 (pitch 136 hw, 69632B), W2 @69632 (pitch 72 hw, 18432B),
//             per-wave 4352B scratch @88064. Total 122880B dynamic.
// ---------------------------------------------------------------------------
#define E_W2_OFF  69632
#define E_SCR_OFF 88064
#define E_LDS     122880

__global__ __launch_bounds__(256) void edge_kernel(
    const int* __restrict__ srcI, const int* __restrict__ dstI,
    const int* __restrict__ n2g, const unsigned short* __restrict__ zpb,
    const unsigned short* __restrict__ w1bf, const unsigned short* __restrict__ w2bf,
    const float* __restrict__ b1, const float* __restrict__ g1v, const float* __restrict__ be1,
    const float* __restrict__ b2, const float* __restrict__ g2v, const float* __restrict__ be2,
    const float* __restrict__ w3, const float* __restrict__ b3,
    float* __restrict__ outE) {
  extern __shared__ unsigned char smem[];
  unsigned short* W1 = (unsigned short*)smem;
  unsigned short* W2 = (unsigned short*)(smem + E_W2_OFF);
  const int lane = threadIdx.x & 31;
  const int wv = threadIdx.x >> 5;
  unsigned short* SCR = (unsigned short*)(smem + E_SCR_OFF + wv * 4352);
  float* SCRF = (float*)SCR;
  const int n15 = lane & 15;
  const int hf = lane >> 4;

  if (threadIdx.x < 32) {  // wave 0 issues the TDM weight loads
    tdm_load2d(0u, w1bf, 128u, 256u, 5u, 3u);
    tdm_load2d((unsigned)E_W2_OFF, w2bf, 64u, 128u, 4u, 3u);
    __builtin_amdgcn_s_wait_tensorcnt(0);
  }
  __syncthreads();

  float bias1[8], gam1[8], bet1[8];
  #pragma unroll
  for (int nt = 0; nt < 8; ++nt) {
    int col = nt * 16 + n15;
    bias1[nt] = b1[col]; gam1[nt] = g1v[col]; bet1[nt] = be1[col];
  }
  float bias2[4], gam2[4], bet2[4];
  #pragma unroll
  for (int nt = 0; nt < 4; ++nt) {
    int col = nt * 16 + n15;
    bias2[nt] = b2[col]; gam2[nt] = g2v[col]; bet2[nt] = be2[col];
  }

  const int nWaves = (int)(gridDim.x * (blockDim.x >> 5));
  int wgid = (int)blockIdx.x * (int)(blockDim.x >> 5) + wv;
  for (int t = wgid; t < NEDGE / 16; t += nWaves) {
    const int ebase = t * 16;
    __builtin_prefetch(srcI + (size_t)(t + nWaves) * 16, 0, 0);
    int e = ebase + n15;
    int gs = n2g[srcI[e]];
    int gd = n2g[dstI[e]];
    const unsigned short* rs = zpb + (size_t)gs * 128;
    const unsigned short* rd = zpb + (size_t)gd * 128;

    v8f acc[8] = {};
    #pragma unroll
    for (int c = 0; c < 8; ++c) {
      const unsigned short* rp = (c < 4) ? (rs + 32 * c) : (rd + (32 * c - 128));
      FragB A;
      A.q[0] = *(const v4u_*)(rp + 8 * hf);
      A.q[1] = *(const v4u_*)(rp + 16 + 8 * hf);
      const unsigned short* wrow = W1 + (unsigned)(32 * c + 8 * hf) * 136u + n15;
      #pragma unroll
      for (int nt = 0; nt < 8; ++nt) {
        FragB Bf;
        #pragma unroll
        for (int s = 0; s < 16; ++s) Bf.s[s] = wrow[((s & 7) + ((s >> 3) << 4)) * 136 + nt * 16];
        acc[nt] = __builtin_amdgcn_wmma_f32_16x16x32_bf16(
            false, A.v, false, Bf.v, (short)0, acc[nt], false, false);
      }
    }
    // bias + GELU + LN over 128 (row M = r + 8*hf)
    float s1[8], s2[8];
    #pragma unroll
    for (int r = 0; r < 8; ++r) { s1[r] = 0.f; s2[r] = 0.f; }
    #pragma unroll
    for (int nt = 0; nt < 8; ++nt) {
      #pragma unroll
      for (int r = 0; r < 8; ++r) {
        float x = gelu_f(acc[nt][r] + bias1[nt]);
        acc[nt][r] = x;
        s1[r] += x; s2[r] += x * x;
      }
    }
    #pragma unroll
    for (int r = 0; r < 8; ++r) {
      #pragma unroll
      for (int m = 1; m < 16; m <<= 1) {
        s1[r] += __shfl_xor(s1[r], m);
        s2[r] += __shfl_xor(s2[r], m);
      }
      float mu = s1[r] * (1.0f / 128.0f);
      float var = s2[r] * (1.0f / 128.0f) - mu * mu;
      float inv = rsqrtf(var + 1e-5f);
      int M = r + 8 * hf;
      #pragma unroll
      for (int nt = 0; nt < 8; ++nt) {
        float y = (acc[nt][r] - mu) * inv * gam1[nt] + bet1[nt];
        SCR[M * 136 + nt * 16 + n15] = f2bf(y);
      }
    }
    // layer 2: [16,128] @ [128,64]
    v8f acc2[4] = {};
    #pragma unroll
    for (int c = 0; c < 4; ++c) {
      const unsigned short* rp = SCR + n15 * 136 + 32 * c;
      FragB A;
      A.q[0] = *(const v4u_*)(rp + 8 * hf);
      A.q[1] = *(const v4u_*)(rp + 16 + 8 * hf);
      const unsigned short* wrow = W2 + (unsigned)(32 * c + 8 * hf) * 72u + n15;
      #pragma unroll
      for (int nt = 0; nt < 4; ++nt) {
        FragB Bf;
        #pragma unroll
        for (int s = 0; s < 16; ++s) Bf.s[s] = wrow[((s & 7) + ((s >> 3) << 4)) * 72 + nt * 16];
        acc2[nt] = __builtin_amdgcn_wmma_f32_16x16x32_bf16(
            false, A.v, false, Bf.v, (short)0, acc2[nt], false, false);
      }
    }
    #pragma unroll
    for (int r = 0; r < 8; ++r) { s1[r] = 0.f; s2[r] = 0.f; }
    #pragma unroll
    for (int nt = 0; nt < 4; ++nt) {
      #pragma unroll
      for (int r = 0; r < 8; ++r) {
        float x = gelu_f(acc2[nt][r] + bias2[nt]);
        acc2[nt][r] = x;
        s1[r] += x; s2[r] += x * x;
      }
    }
    #pragma unroll
    for (int r = 0; r < 8; ++r) {
      #pragma unroll
      for (int m = 1; m < 16; m <<= 1) {
        s1[r] += __shfl_xor(s1[r], m);
        s2[r] += __shfl_xor(s2[r], m);
      }
      float mu = s1[r] * (1.0f / 64.0f);
      float var = s2[r] * (1.0f / 64.0f) - mu * mu;
      float inv = rsqrtf(var + 1e-5f);
      int M = r + 8 * hf;
      #pragma unroll
      for (int nt = 0; nt < 4; ++nt) {
        float y = (acc2[nt][r] - mu) * inv * gam2[nt] + bet2[nt];
        SCRF[M * 68 + nt * 16 + n15] = y;
      }
    }
    // final [16,64] @ [64,3] + b3
    for (int tt = lane; tt < 48; tt += 32) {
      int m = tt / 3, j = tt - m * 3;
      float a = b3[j];
      const float* rowp = SCRF + m * 68;
      #pragma unroll
      for (int k = 0; k < 64; ++k) a += rowp[k] * w3[k * 3 + j];
      outE[(size_t)(ebase + m) * 3 + j] = a;
    }
  }
}

// ---------------------------------------------------------------------------
// NODE kernel: per wave, a tile of 16 nodes.
//   L1: [16,128](node_emb f32->bf16) @ [128,128](LDS) -> GELU+LN -> +z_proj[g]
//   L2: [16,128] @ [128,64] -> GELU+LN -> [64,4] proj
// LDS: ne_w @0 (pitch 136, 34816B), nd_w1 @34816 (pitch 72, 18432B),
//      per-wave 4352B scratch @53248. Total 88064B dynamic.
// ---------------------------------------------------------------------------
#define N_ND1_OFF 34816
#define N_SCR_OFF 53248
#define N_LDS     88064

__device__ __forceinline__ void pack8(const float* p, unsigned short* s) {
  float4 a = *(const float4*)p;
  float4 b = *(const float4*)(p + 4);
  s[0] = f2bf(a.x); s[1] = f2bf(a.y); s[2] = f2bf(a.z); s[3] = f2bf(a.w);
  s[4] = f2bf(b.x); s[5] = f2bf(b.y); s[6] = f2bf(b.z); s[7] = f2bf(b.w);
}

__global__ __launch_bounds__(256) void node_kernel(
    const float* __restrict__ nemb, const int* __restrict__ n2g,
    const float* __restrict__ zpf,
    const unsigned short* __restrict__ newbf, const unsigned short* __restrict__ nd1bf,
    const float* __restrict__ neb, const float* __restrict__ negv, const float* __restrict__ nebe,
    const float* __restrict__ ndb1, const float* __restrict__ ndg, const float* __restrict__ ndbe,
    const float* __restrict__ ndw2, const float* __restrict__ ndb2,
    float* __restrict__ outN) {
  extern __shared__ unsigned char smem[];
  unsigned short* WA = (unsigned short*)smem;
  unsigned short* WB = (unsigned short*)(smem + N_ND1_OFF);
  const int lane = threadIdx.x & 31;
  const int wv = threadIdx.x >> 5;
  unsigned short* SCR = (unsigned short*)(smem + N_SCR_OFF + wv * 4352);
  float* SCRF = (float*)SCR;
  const int n15 = lane & 15;
  const int hf = lane >> 4;

  if (threadIdx.x < 32) {
    tdm_load2d(0u, newbf, 128u, 128u, 5u, 3u);
    tdm_load2d((unsigned)N_ND1_OFF, nd1bf, 64u, 128u, 4u, 3u);
    __builtin_amdgcn_s_wait_tensorcnt(0);
  }
  __syncthreads();

  float bias1[8], gam1[8], bet1[8];
  #pragma unroll
  for (int nt = 0; nt < 8; ++nt) {
    int col = nt * 16 + n15;
    bias1[nt] = neb[col]; gam1[nt] = negv[col]; bet1[nt] = nebe[col];
  }
  float bias2[4], gam2[4], bet2[4];
  #pragma unroll
  for (int nt = 0; nt < 4; ++nt) {
    int col = nt * 16 + n15;
    bias2[nt] = ndb1[col]; gam2[nt] = ndg[col]; bet2[nt] = ndbe[col];
  }

  const int nWaves = (int)(gridDim.x * (blockDim.x >> 5));
  int wgid = (int)blockIdx.x * (int)(blockDim.x >> 5) + wv;
  for (int t = wgid; t < NNODE / 16; t += nWaves) {
    const int nb = t * 16;
    int node = nb + n15;
    int g = n2g[node];
    const float* nr = nemb + (size_t)node * 128;
    __builtin_prefetch(nemb + (size_t)(nb + nWaves * 16) * 128, 0, 0);

    v8f acc[8] = {};
    #pragma unroll
    for (int c = 0; c < 4; ++c) {
      FragB A;
      const float* p0 = nr + 32 * c + 8 * hf;
      pack8(p0, A.s);
      pack8(p0 + 16, A.s + 8);
      const unsigned short* wrow = WA + (unsigned)(32 * c + 8 * hf) * 136u + n15;
      #pragma unroll
      for (int nt = 0; nt < 8; ++nt) {
        FragB Bf;
        #pragma unroll
        for (int s = 0; s < 16; ++s) Bf.s[s] = wrow[((s & 7) + ((s >> 3) << 4)) * 136 + nt * 16];
        acc[nt] = __builtin_amdgcn_wmma_f32_16x16x32_bf16(
            false, A.v, false, Bf.v, (short)0, acc[nt], false, false);
      }
    }
    // bias + GELU + LN(128), then add z_proj[graph] -> bf16 stage
    float s1[8], s2[8];
    #pragma unroll
    for (int r = 0; r < 8; ++r) { s1[r] = 0.f; s2[r] = 0.f; }
    #pragma unroll
    for (int nt = 0; nt < 8; ++nt) {
      #pragma unroll
      for (int r = 0; r < 8; ++r) {
        float x = gelu_f(acc[nt][r] + bias1[nt]);
        acc[nt][r] = x;
        s1[r] += x; s2[r] += x * x;
      }
    }
    #pragma unroll
    for (int r = 0; r < 8; ++r) {
      #pragma unroll
      for (int m = 1; m < 16; m <<= 1) {
        s1[r] += __shfl_xor(s1[r], m);
        s2[r] += __shfl_xor(s2[r], m);
      }
      float mu = s1[r] * (1.0f / 128.0f);
      float var = s2[r] * (1.0f / 128.0f) - mu * mu;
      float inv = rsqrtf(var + 1e-5f);
      int M = r + 8 * hf;
      int gm = __shfl(g, M);
      const float* zrow = zpf + (size_t)gm * 128;
      #pragma unroll
      for (int nt = 0; nt < 8; ++nt) {
        float y = (acc[nt][r] - mu) * inv * gam1[nt] + bet1[nt];
        y += zrow[nt * 16 + n15];
        SCR[M * 136 + nt * 16 + n15] = f2bf(y);
      }
    }
    // layer 2: [16,128] @ [128,64]
    v8f acc2[4] = {};
    #pragma unroll
    for (int c = 0; c < 4; ++c) {
      const unsigned short* rp = SCR + n15 * 136 + 32 * c;
      FragB A;
      A.q[0] = *(const v4u_*)(rp + 8 * hf);
      A.q[1] = *(const v4u_*)(rp + 16 + 8 * hf);
      const unsigned short* wrow = WB + (unsigned)(32 * c + 8 * hf) * 72u + n15;
      #pragma unroll
      for (int nt = 0; nt < 4; ++nt) {
        FragB Bf;
        #pragma unroll
        for (int s = 0; s < 16; ++s) Bf.s[s] = wrow[((s & 7) + ((s >> 3) << 4)) * 72 + nt * 16];
        acc2[nt] = __builtin_amdgcn_wmma_f32_16x16x32_bf16(
            false, A.v, false, Bf.v, (short)0, acc2[nt], false, false);
      }
    }
    #pragma unroll
    for (int r = 0; r < 8; ++r) { s1[r] = 0.f; s2[r] = 0.f; }
    #pragma unroll
    for (int nt = 0; nt < 4; ++nt) {
      #pragma unroll
      for (int r = 0; r < 8; ++r) {
        float x = gelu_f(acc2[nt][r] + bias2[nt]);
        acc2[nt][r] = x;
        s1[r] += x; s2[r] += x * x;
      }
    }
    #pragma unroll
    for (int r = 0; r < 8; ++r) {
      #pragma unroll
      for (int m = 1; m < 16; m <<= 1) {
        s1[r] += __shfl_xor(s1[r], m);
        s2[r] += __shfl_xor(s2[r], m);
      }
      float mu = s1[r] * (1.0f / 64.0f);
      float var = s2[r] * (1.0f / 64.0f) - mu * mu;
      float inv = rsqrtf(var + 1e-5f);
      int M = r + 8 * hf;
      #pragma unroll
      for (int nt = 0; nt < 4; ++nt) {
        float y = (acc2[nt][r] - mu) * inv * gam2[nt] + bet2[nt];
        SCRF[M * 68 + nt * 16 + n15] = y;
      }
    }
    // final [16,64] @ [64,4] + b2
    #pragma unroll
    for (int it = 0; it < 2; ++it) {
      int tt = lane + it * 32;
      int m = tt >> 2, j = tt & 3;
      float a = ndb2[j];
      const float* rowp = SCRF + m * 68;
      #pragma unroll
      for (int k = 0; k < 64; ++k) a += rowp[k] * ndw2[k * 4 + j];
      outN[(size_t)(nb + m) * 4 + j] = a;
    }
  }
}

// ---------------------------------------------------------------------------
extern "C" void kernel_launch(void* const* d_in, const int* in_sizes, int n_in,
                              void* d_out, int out_size, void* d_ws, size_t ws_size,
                              hipStream_t stream) {
  (void)in_sizes; (void)n_in; (void)out_size; (void)ws_size;
  const float* z        = (const float*)d_in[0];
  const float* node_emb = (const float*)d_in[1];
  const int*   n2g      = (const int*)d_in[2];
  const int*   srcI     = (const int*)d_in[3];
  const int*   dstI     = (const int*)d_in[4];
  const float* lp_w  = (const float*)d_in[5],  *lp_b  = (const float*)d_in[6];
  const float* lp_g  = (const float*)d_in[7],  *lp_be = (const float*)d_in[8];
  const float* ne_w  = (const float*)d_in[9],  *ne_b  = (const float*)d_in[10];
  const float* ne_g  = (const float*)d_in[11], *ne_be = (const float*)d_in[12];
  const float* nd_w1 = (const float*)d_in[13], *nd_b1 = (const float*)d_in[14];
  const float* nd_g  = (const float*)d_in[15], *nd_be = (const float*)d_in[16];
  const float* nd_w2 = (const float*)d_in[17], *nd_b2 = (const float*)d_in[18];
  const float* ed_w1 = (const float*)d_in[19], *ed_b1 = (const float*)d_in[20];
  const float* ed_g1 = (const float*)d_in[21], *ed_be1= (const float*)d_in[22];
  const float* ed_w2 = (const float*)d_in[23], *ed_b2 = (const float*)d_in[24];
  const float* ed_g2 = (const float*)d_in[25], *ed_be2= (const float*)d_in[26];
  const float* ed_w3 = (const float*)d_in[27], *ed_b3 = (const float*)d_in[28];
  const float* en_w1 = (const float*)d_in[29], *en_b1 = (const float*)d_in[30];
  const float* en_g  = (const float*)d_in[31], *en_be = (const float*)d_in[32];
  const float* en_w2 = (const float*)d_in[33], *en_b2 = (const float*)d_in[34];
  const float* st_w1 = (const float*)d_in[35], *st_b1 = (const float*)d_in[36];
  const float* st_g  = (const float*)d_in[37], *st_be = (const float*)d_in[38];
  const float* st_w2 = (const float*)d_in[39], *st_b2 = (const float*)d_in[40];

  float* out = (float*)d_out;
  char* ws = (char*)d_ws;
  float*          zpf   = (float*)(ws + 0);                 // 1024*128 f32
  unsigned short* zpb   = (unsigned short*)(ws + 524288);   // 1024*128 bf16
  unsigned short* w1bf  = (unsigned short*)(ws + 786432);   // 256*128 bf16
  unsigned short* w2bf  = (unsigned short*)(ws + 851968);   // 128*64 bf16
  unsigned short* newbf = (unsigned short*)(ws + 868352);   // 128*128 bf16
  unsigned short* nd1bf = (unsigned short*)(ws + 901120);   // 128*64 bf16

  convert_kernel<<<128, 256, 0, stream>>>(ed_w1, ed_w2, ne_w, nd_w1,
                                          w1bf, w2bf, newbf, nd1bf);
  zproj_kernel<<<NB, 128, 0, stream>>>(z, lp_w, lp_b, lp_g, lp_be, zpf, zpb);
  heads_kernel<<<NB, 64, 0, stream>>>(z, en_w1, en_b1, en_g, en_be, en_w2, en_b2,
                                      st_w1, st_b1, st_g, st_be, st_w2, st_b2,
                                      out + 3819200, out + 3821248);
  node_kernel<<<512, 256, N_LDS, stream>>>(node_emb, n2g, zpf, newbf, nd1bf,
                                           ne_b, ne_g, ne_be, nd_b1, nd_g, nd_be,
                                           nd_w2, nd_b2, out);
  edge_kernel<<<1024, 256, E_LDS, stream>>>(srcI, dstI, n2g, zpb, w1bf, w2bf,
                                            ed_b1, ed_g1, ed_be1, ed_b2, ed_g2, ed_be2,
                                            ed_w3, ed_b3, out + 819200);
}
